// MHGNN_66005057405231
// MI455X (gfx1250) — matrix operand order repaired and checked by reading, not compiled, not run on hardware
//
#include <hip/hip_runtime.h>
#include <hip/hip_bf16.h>
#include <math.h>

typedef __attribute__((ext_vector_type(16))) _Float16 v16h;
typedef __attribute__((ext_vector_type(8)))  float    v8f;

#define HID     64
#define NPG     1000
#define NGRAPH  100
#define KPG     800

static inline int cdiv_i(long long a, long long b) { return (int)((a + b - 1) / b); }

// ---------------- degree / dinv ----------------
__global__ void k_count_deg(const int* __restrict__ dst, float* __restrict__ deg, int E) {
    int e = blockIdx.x * blockDim.x + threadIdx.x;
    if (e < E) atomicAdd(&deg[dst[e]], 1.0f);
}

__global__ void k_dinv(float* __restrict__ deg, int N) {
    int n = blockIdx.x * blockDim.x + threadIdx.x;
    if (n < N) deg[n] = rsqrtf(deg[n] + 1.0f);   // +1 self-loop
}

// ---------------- agg init: self-loop term + bias ----------------
__global__ void k_init_agg(const float* __restrict__ H, const float* __restrict__ dinv,
                           const float* __restrict__ b, float* __restrict__ agg, int Nn) {
    int t = blockIdx.x * blockDim.x + threadIdx.x;
    if (t < Nn * HID) {
        int n = t >> 6, f = t & 63;
        float d = dinv[n];
        agg[t] = H[t] * d * d + b[f];
    }
}

// ---------------- edge scatter (one thread per edge-feature) ----------------
__global__ void k_scatter(const int* __restrict__ src, const int* __restrict__ dst,
                          const float* __restrict__ dinv, const float* __restrict__ H,
                          float* __restrict__ agg, int E) {
    int t = blockIdx.x * blockDim.x + threadIdx.x;
    int e = t >> 6, f = t & 63;
    if (e < E) {
        int s = src[e], d = dst[e];
        float nrm = dinv[s] * dinv[d];
        atomicAdd(&agg[(size_t)d * HID + f], H[(size_t)s * HID + f] * nrm);
    }
}

// ---------------- pooling score: relu(h).pool_w / ||pool_w|| ----------------
__global__ void k_score(const float* __restrict__ agg, const float* __restrict__ pw,
                        float* __restrict__ score, int Nn) {
    int n = blockIdx.x * blockDim.x + threadIdx.x;
    if (n >= Nn) return;
    float w2 = 0.f;
#pragma unroll
    for (int f = 0; f < HID; ++f) w2 += pw[f] * pw[f];
    const float* row = agg + (size_t)n * HID;
    float s = 0.f;
#pragma unroll
    for (int f = 0; f < HID; ++f) s += fmaxf(row[f], 0.f) * pw[f];
    score[n] = s * rsqrtf(w2);
}

// ---------------- per-graph sorted top-k via bitonic sort in LDS ----------------
__global__ __launch_bounds__(1024)
void k_topk(const float* __restrict__ score, int* __restrict__ perm, float* __restrict__ tsc) {
    __shared__ float s[1024];
    __shared__ int   id[1024];
    int g = blockIdx.x;
    int t = threadIdx.x;
    if (t < NPG) { s[t] = score[g * NPG + t]; id[t] = t; }
    else         { s[t] = -INFINITY;          id[t] = 1 << 30; }
    __syncthreads();
    for (int k = 2; k <= 1024; k <<= 1) {
        for (int j = k >> 1; j > 0; j >>= 1) {
            int ixj = t ^ j;
            if (ixj > t) {
                float sa = s[t], sb = s[ixj];
                int   ia = id[t], ib = id[ixj];
                bool up = ((t & k) == 0);
                // "before" = higher score first; ties -> lower index (matches lax.top_k)
                bool b_before_a = (sb > sa) || (sb == sa && ib < ia);
                bool a_before_b = (sa > sb) || (sa == sb && ia < ib);
                bool sw = up ? b_before_a : a_before_b;
                if (sw) { s[t] = sb; s[ixj] = sa; id[t] = ib; id[ixj] = ia; }
            }
            __syncthreads();
        }
    }
    if (t < KPG) {
        perm[g * KPG + t] = g * NPG + id[t];
        tsc[g * KPG + t]  = tanhf(s[t]);
    }
}

// ---------------- product gather: p[i] = relu(h[perm[pidx[i]]]) * tanh(score) ----------------
__global__ void k_gather(const int* __restrict__ pidx, const int* __restrict__ perm,
                         const float* __restrict__ tsc, const float* __restrict__ agg,
                         float* __restrict__ P, int NP) {
    int t = blockIdx.x * blockDim.x + threadIdx.x;
    int i = t >> 6, f = t & 63;
    if (i < NP) {
        int pi = pidx[i];
        int node = perm[pi];
        float ts = tsc[pi];
        P[(size_t)i * HID + f] = fmaxf(agg[(size_t)node * HID + f], 0.f) * ts;
    }
}

// ---------------- WMMA GEMM: C[M,N] = (relu?)(A[M,K]) @ W[K,N] (+bias) ----------------
// Weights staged once per block into LDS as f16, transposed to [col][k] with a
// +8-half row pad (stride 272B -> bank step 4 -> conflict-free b128 fragment reads).
// Inner loop: all NT B-fragments are fetched from LDS *before* the WMMA burst so
// the compiler can clause the ds_loads behind one s_wait_dscnt, then issue NT
// back-to-back v_wmma_f32_16x16x32_f16.
// 256-thread block = 8 waves = 128 rows. M is always a multiple of 16 here.
template<int NT, int KK, bool RELU_A, bool HAS_BIAS>
__global__ __launch_bounds__(256)
void k_gemm(const float* __restrict__ A, const float* __restrict__ W,
            const float* __restrict__ bias, float* __restrict__ C,
            int M, int N) {
    constexpr int NC  = NT * 16;     // padded column count
    constexpr int KKP = KK + 8;      // padded LDS row stride (halves)
    __shared__ _Float16 Wl[NC * KKP];

    // ---- stage W -> LDS (f16, transposed, zero-padded cols) ----
    for (int idx = threadIdx.x; idx < NC * KK; idx += 256) {
        int c = idx / KK;            // output column
        int k = idx % KK;            // contiguous in LDS: conflict-free writes
        float wv = (c < N) ? W[(size_t)k * N + c] : 0.f;
        Wl[c * KKP + k] = (_Float16)wv;
    }
    __syncthreads();

    int lane = threadIdx.x & 31;
    int wv   = threadIdx.x >> 5;
    int row0 = (blockIdx.x * 8 + wv) * 16;
    if (row0 >= M) return;           // after the barrier; wave-uniform -> EXEC all-ones
    int m  = lane & 15;              // A: row within tile; B/C: column within tile
    int kh = lane >> 4;              // lane half selects K sub-block / C row half

    v8f acc[NT];
#pragma unroll
    for (int ct = 0; ct < NT; ++ct) { v8f z = {}; acc[ct] = z; }

    union BF { float4 f4[2]; v16h v; };

#pragma unroll
    for (int k0 = 0; k0 < KK; k0 += 32) {
        // ---- A fragment (16x32 f16): elements 0..7 -> K=k0+kh*8+0..7,
        //                              elements 8..15 -> K=k0+16+kh*8+0..7
        const float* ap = A + (size_t)(row0 + m) * KK + k0 + kh * 8;
        float4 a0 = *(const float4*)(ap);
        float4 a1 = *(const float4*)(ap + 4);
        float4 a2 = *(const float4*)(ap + 16);
        float4 a3 = *(const float4*)(ap + 20);
        float af[16] = {a0.x, a0.y, a0.z, a0.w, a1.x, a1.y, a1.z, a1.w,
                        a2.x, a2.y, a2.z, a2.w, a3.x, a3.y, a3.z, a3.w};
        v16h av;
#pragma unroll
        for (int e = 0; e < 16; ++e) {
            float v = af[e];
            if (RELU_A) v = fmaxf(v, 0.f);
            av[e] = (_Float16)v;
        }

        // ---- fetch ALL B fragments first (one ds_load clause, one wait) ----
        int kb = k0 + kh * 8;
        BF b[NT];
#pragma unroll
        for (int ct = 0; ct < NT; ++ct) {
            const _Float16* bp = &Wl[(ct * 16 + m) * KKP + kb];
            b[ct].f4[0] = *(const float4*)(bp);
            b[ct].f4[1] = *(const float4*)(bp + 16);
        }

        // ---- WMMA burst ----
#pragma unroll
        for (int ct = 0; ct < NT; ++ct) {
            acc[ct] = __builtin_amdgcn_wmma_f32_16x16x32_f16(
                false, av, false, b[ct].v, (short)0, acc[ct], false, false);
        }
    }

    // ---- store: acc element r -> row row0 + kh*8 + r, col = ct*16 + m
#pragma unroll
    for (int ct = 0; ct < NT; ++ct) {
        int col = ct * 16 + m;
        if (col < N) {
            float bb = HAS_BIAS ? bias[col] : 0.f;
#pragma unroll
            for (int r = 0; r < 8; ++r) {
                int row = row0 + kh * 8 + r;
                C[(size_t)row * N + col] = acc[ct][r] + bb;
            }
        }
    }
}

// =====================================================================
extern "C" void kernel_launch(void* const* d_in, const int* in_sizes, int n_in,
                              void* d_out, int out_size, void* d_ws, size_t ws_size,
                              hipStream_t stream) {
    const float* x    = (const float*)d_in[0];
    const int*   ei   = (const int*)  d_in[1];
    const int*   batch= (const int*)  d_in[2]; (void)batch; // equal-size graphs
    const int*   pidx = (const int*)  d_in[3];
    const float* W1   = (const float*)d_in[4];
    const float* b1   = (const float*)d_in[5];
    const float* W2   = (const float*)d_in[6];
    const float* b2   = (const float*)d_in[7];
    const float* pw   = (const float*)d_in[8];
    const float* fc1W = (const float*)d_in[9];
    const float* fc1b = (const float*)d_in[10];
    const float* fc2W = (const float*)d_in[11];
    const float* fc2b = (const float*)d_in[12];
    float* out = (float*)d_out;
    (void)n_in; (void)out_size; (void)ws_size;

    const int Nn = in_sizes[0] / 128;   // 100000 nodes
    const int E  = in_sizes[1] / 2;     // 1600000 edges
    const int NP = in_sizes[3];         // 20000 product rows
    const int* src = ei;
    const int* dst = ei + E;

    // ---- workspace carve (256B aligned) ----
    char* wp = (char*)d_ws;
    auto carve = [&](size_t bytes) -> char* {
        char* p = wp; wp += (bytes + 255) & ~(size_t)255; return p;
    };
    float* dinv = (float*)carve((size_t)Nn * 4);               // deg -> dinv -> score (reused)
    float* H    = (float*)carve((size_t)Nn * HID * 4);         // h1 -> h2 -> P (reused)
    float* AG   = (float*)carve((size_t)Nn * HID * 4);         // agg1 -> agg2 -> F1 (reused)
    int*   perm = (int*)  carve((size_t)NGRAPH * KPG * 4);
    float* tsc  = (float*)carve((size_t)NGRAPH * KPG * 4);
    float* P  = H;    // 20000x64, H no longer needed by then
    float* F1 = AG;   // 20000x64, AG no longer needed by then

    const int T = 256;

    // 1) degree -> dinv
    hipMemsetAsync(dinv, 0, (size_t)Nn * 4, stream);
    k_count_deg<<<cdiv_i(E, T), T, 0, stream>>>(dst, dinv, E);
    k_dinv<<<cdiv_i(Nn, T), T, 0, stream>>>(dinv, Nn);

    // 2) GCN layer 1: H = x @ W1 ; AG = H*dinv^2 + b1 ; AG += scatter(H*norm)
    k_gemm<4, 128, false, false><<<cdiv_i((long long)Nn, 128), T, 0, stream>>>(x, W1, nullptr, H, Nn, HID);
    k_init_agg<<<cdiv_i((long long)Nn * HID, T), T, 0, stream>>>(H, dinv, b1, AG, Nn);
    k_scatter<<<cdiv_i((long long)E * HID, T), T, 0, stream>>>(src, dst, dinv, H, AG, E);

    // 3) GCN layer 2 (ReLU fused into A-load): H = relu(AG) @ W2 ; AG = H*dinv^2 + b2 ; scatter
    k_gemm<4, 64, true, false><<<cdiv_i((long long)Nn, 128), T, 0, stream>>>(AG, W2, nullptr, H, Nn, HID);
    k_init_agg<<<cdiv_i((long long)Nn * HID, T), T, 0, stream>>>(H, dinv, b2, AG, Nn);
    k_scatter<<<cdiv_i((long long)E * HID, T), T, 0, stream>>>(src, dst, dinv, H, AG, E);

    // 4) TopK pooling: score = relu(AG).pw/||pw|| ; per-graph sorted top-800
    k_score<<<cdiv_i(Nn, T), T, 0, stream>>>(AG, pw, dinv, Nn);   // dinv reused as score buf
    k_topk<<<NGRAPH, 1024, 0, stream>>>(dinv, perm, tsc);

    // 5) product gather (pooled matrix never materialized)
    k_gather<<<cdiv_i((long long)NP * HID, T), T, 0, stream>>>(pidx, perm, tsc, AG, P, NP);

    // 6) MLP head: F1 = P@fc1W + fc1b ; out = relu(F1)@fc2W + fc2b  (N=HID, one 16-col tile)
    k_gemm<4, 64, false, true><<<cdiv_i((long long)NP, 128), T, 0, stream>>>(P, fc1W, fc1b, F1, NP, HID);
    k_gemm<1, 64, true,  true><<<cdiv_i((long long)NP, 128), T, 0, stream>>>(F1, fc2W, fc2b, out, NP, HID);
}
